// Qwen3NextSparseMoeBlock_15083925143801
// MI455X (gfx1250) — compile-verified, matrix operand order checked by reading, not run on hardware
//
#include <hip/hip_runtime.h>
#include <math.h>

// ---------------- problem constants ----------------
#define TT 4096
#define HH 1024
#define EE 16
#define II 512
#define SSI 2048
#define MAXROWS 8448            // 2*T + worst-case per-expert padding (<=15*16)
#define MAXTILES (MAXROWS / 16) // 528

// ---------------- WMMA types ----------------
typedef __attribute__((ext_vector_type(16))) __bf16 v16bf;
typedef __attribute__((ext_vector_type(8)))  __bf16 v8bf;
typedef __attribute__((ext_vector_type(4)))  __bf16 v4bf;
typedef __attribute__((ext_vector_type(8)))  float  v8f;

// GEMM tiling
#define BM 16
#define BN 128
#define BK 32
#define BKP 40   // padded LDS K-stride (elements): 80B row stride, 16B aligned, conflict-free

// Fragment load: 16-bit A/B layout = two contiguous 8-element K-runs per lane:
//   run0 at K = (lane>>4)*8, run1 at K = 16 + (lane>>4)*8   (ISA 7.12.2)
__device__ __forceinline__ v16bf ld_frag(const __bf16* rowbase, int lane) {
  int ko = (lane >> 4) * 8;
  v8bf lo = *(const v8bf*)(rowbase + ko);
  v8bf hi = *(const v8bf*)(rowbase + 16 + ko);
  v16bf r;
#pragma unroll
  for (int i = 0; i < 8; ++i) { r[i] = lo[i]; r[i + 8] = hi[i]; }
  return r;
}

__device__ __forceinline__ float fast_sigmoid(float x) {
  return __builtin_amdgcn_rcpf(1.0f + __expf(-x));
}
__device__ __forceinline__ float silu(float x) { return x * fast_sigmoid(x); }

// pack 4 floats -> v4bf
__device__ __forceinline__ v4bf pk4(const float* f) {
  v4bf o;
  o[0] = (__bf16)f[0]; o[1] = (__bf16)f[1]; o[2] = (__bf16)f[2]; o[3] = (__bf16)f[3];
  return o;
}

// ---------------- small kernels ----------------
__global__ void cvt_bf16_kernel(const float* __restrict__ x, __bf16* __restrict__ xb, int n4) {
  int i = (blockIdx.x * blockDim.x + threadIdx.x);
  if (i < n4) {
    float4 v = ((const float4*)x)[i];
    float f[4] = {v.x, v.y, v.z, v.w};
    ((v4bf*)xb)[i] = pk4(f);
  }
}

__global__ void init_kernel(int* __restrict__ counts, int* __restrict__ cursor) {
  int i = threadIdx.x;
  if (i < EE) { counts[i] = 0; cursor[i] = 0; }
}

// one wave per token: logits (E=16 dots of len H), softmax, top-2 renorm, shared sigmoid gate
__global__ __launch_bounds__(256)
void router_kernel(const float* __restrict__ x, const float* __restrict__ wr,
                   const float* __restrict__ wsg,
                   int* __restrict__ topi, float* __restrict__ topw, float* __restrict__ sg) {
  int wave = threadIdx.x >> 5;
  int lane = threadIdx.x & 31;
  int t = blockIdx.x * 8 + wave;
  const float* xr = x + (size_t)t * HH;
  float acc[EE];
#pragma unroll
  for (int e = 0; e < EE; ++e) acc[e] = 0.f;
  float asg = 0.f;
  for (int h = lane; h < HH; h += 32) {
    float xv = xr[h];
    asg += xv * wsg[h];
    const float* wrow = wr + (size_t)h * EE;
#pragma unroll
    for (int e = 0; e < EE; ++e) acc[e] += xv * wrow[e];
  }
#pragma unroll
  for (int off = 16; off > 0; off >>= 1) {
#pragma unroll
    for (int e = 0; e < EE; ++e) acc[e] += __shfl_xor(acc[e], off, 32);
    asg += __shfl_xor(asg, off, 32);
  }
  if (lane == 0) {
    float m = acc[0];
#pragma unroll
    for (int e = 1; e < EE; ++e) m = fmaxf(m, acc[e]);
    float p[EE], s = 0.f;
#pragma unroll
    for (int e = 0; e < EE; ++e) { p[e] = __expf(acc[e] - m); s += p[e]; }
    float inv = __builtin_amdgcn_rcpf(s);
    int i0 = 0; float p0 = -1.f;
#pragma unroll
    for (int e = 0; e < EE; ++e) { float v = p[e] * inv; p[e] = v; if (v > p0) { p0 = v; i0 = e; } }
    int i1 = 0; float p1 = -1.f;
#pragma unroll
    for (int e = 0; e < EE; ++e) { if (e != i0 && p[e] > p1) { p1 = p[e]; i1 = e; } }
    float rden = __builtin_amdgcn_rcpf(p0 + p1);
    topi[t * 2 + 0] = i0; topi[t * 2 + 1] = i1;
    topw[t * 2 + 0] = p0 * rden; topw[t * 2 + 1] = p1 * rden;
    sg[t] = fast_sigmoid(asg);
  }
}

__global__ void count_kernel(const int* __restrict__ topi, int* __restrict__ counts) {
  int i = blockIdx.x * blockDim.x + threadIdx.x;
  if (i < TT * 2) atomicAdd(&counts[topi[i]], 1);
}

__global__ void offsets_kernel(const int* __restrict__ counts, int* __restrict__ offs,
                               int* __restrict__ rowtok, float* __restrict__ roww) {
  if (threadIdx.x == 0 && blockIdx.x == 0) {
    int off = 0;
    for (int e = 0; e < EE; ++e) {
      offs[e] = off;
      int c = counts[e];
      int p = (c + 15) & ~15;
      for (int i = c; i < p; ++i) { rowtok[off + i] = 0; roww[off + i] = 0.f; }  // zero-weight pad rows
      off += p;
    }
    offs[EE] = off;
  }
}

__global__ void place_kernel(const int* __restrict__ topi, const float* __restrict__ topw,
                             const int* __restrict__ offs, int* __restrict__ cursor,
                             int* __restrict__ rowtok, float* __restrict__ roww) {
  int i = blockIdx.x * blockDim.x + threadIdx.x;
  if (i < TT * 2) {
    int e = topi[i];
    int slot = atomicAdd(&cursor[e], 1);
    rowtok[offs[e] + slot] = i >> 1;
    roww[offs[e] + slot] = topw[i];
  }
}

// ---- staging helpers (register double-buffer pipeline) ----
// B tile: thread owns column nn = tid&127, contiguous K-slice bkh..bkh+15 (bkh = (tid>>7)*16).
// Global loads coalesce along nn; LDS commit = 4x ds_store_b64 of packed bf16.
#define B_LOAD(reg, base, ldb, k0)                                            \
  _Pragma("unroll") for (int r = 0; r < 16; ++r)                              \
      reg[r] = base[(size_t)((k0) + bkh + r) * (ldb) + n0 + bn];
#define B_STORE(reg, Bs)                                                      \
  _Pragma("unroll") for (int q = 0; q < 4; ++q)                               \
      *(v4bf*)(&Bs[bn * BKP + bkh + 4 * q]) = pk4(&reg[4 * q]);

// ---------------- shared expert GEMM1: smid = silu(x@Wg) * (x@Wu), bf16 out ----------------
__global__ __launch_bounds__(256)
void shared_gemm1(const __bf16* __restrict__ xb, const float* __restrict__ wg,
                  const float* __restrict__ wu, __bf16* __restrict__ smid) {
  __shared__ __bf16 As[BM * BKP];
  __shared__ __bf16 Bg[BN * BKP];
  __shared__ __bf16 Bu[BN * BKP];
  int t0 = blockIdx.x * BM;
  int n0 = blockIdx.y * BN;
  int tid = threadIdx.x, lane = tid & 31, wave = tid >> 5;
  int lrow = lane & 15;
  int am = tid >> 4, ak = (tid & 15) * 2;
  int bn = tid & 127, bkh = (tid >> 7) * 16;
  const __bf16* arow = xb + (size_t)(t0 + am) * HH + ak;
  v8f cg = {}; v8f cu = {};
  unsigned aReg; float bgR[16], buR[16];
  aReg = *(const unsigned*)(arow);
  B_LOAD(bgR, wg, SSI, 0)
  B_LOAD(buR, wu, SSI, 0)
  for (int k0 = 0; k0 < HH; k0 += BK) {
    __syncthreads();
    *(unsigned*)(&As[am * BKP + ak]) = aReg;
    B_STORE(bgR, Bg)
    B_STORE(buR, Bu)
    __syncthreads();
    int kn = k0 + BK;
    if (kn < HH) {  // prefetch next tile into registers while WMMAs run
      aReg = *(const unsigned*)(arow + kn);
      B_LOAD(bgR, wg, SSI, kn)
      B_LOAD(buR, wu, SSI, kn)
    }
    v16bf a = ld_frag(&As[lrow * BKP], lane);
    v16bf fg = ld_frag(&Bg[(wave * 16 + lrow) * BKP], lane);
    v16bf fu = ld_frag(&Bu[(wave * 16 + lrow) * BKP], lane);
    cg = __builtin_amdgcn_wmma_f32_16x16x32_bf16(false, a, false, fg, (short)0, cg, false, false);
    cu = __builtin_amdgcn_wmma_f32_16x16x32_bf16(false, a, false, fu, (short)0, cu, false, false);
  }
  int colg = n0 + wave * 16 + (lane & 15);
  int mbase = (lane >> 4) * 8;
#pragma unroll
  for (int r = 0; r < 8; ++r) {
    float h = silu(cg[r]) * cu[r];
    smid[(size_t)(t0 + mbase + r) * SSI + colg] = (__bf16)h;
  }
}

// ---------------- shared expert GEMM2: out = sg[t] * (smid @ Wd), plain store ----------------
__global__ __launch_bounds__(256)
void shared_gemm2(const __bf16* __restrict__ smid, const float* __restrict__ wd,
                  const float* __restrict__ sg, float* __restrict__ out) {
  __shared__ __bf16 As[BM * BKP];
  __shared__ __bf16 Bs[BN * BKP];
  __shared__ float sgs[BM];
  int t0 = blockIdx.x * BM;
  int n0 = blockIdx.y * BN;
  int tid = threadIdx.x, lane = tid & 31, wave = tid >> 5;
  int lrow = lane & 15;
  int am = tid >> 4, ak = (tid & 15) * 2;
  int bn = tid & 127, bkh = (tid >> 7) * 16;
  if (tid < BM) sgs[tid] = sg[t0 + tid];
  const __bf16* arow = smid + (size_t)(t0 + am) * SSI + ak;
  v8f c = {};
  unsigned aReg; float bR[16];
  aReg = *(const unsigned*)(arow);
  B_LOAD(bR, wd, HH, 0)
  for (int k0 = 0; k0 < SSI; k0 += BK) {
    __syncthreads();
    *(unsigned*)(&As[am * BKP + ak]) = aReg;
    B_STORE(bR, Bs)
    __syncthreads();
    int kn = k0 + BK;
    if (kn < SSI) {
      aReg = *(const unsigned*)(arow + kn);
      B_LOAD(bR, wd, HH, kn)
    }
    v16bf a = ld_frag(&As[lrow * BKP], lane);
    v16bf b = ld_frag(&Bs[(wave * 16 + lrow) * BKP], lane);
    c = __builtin_amdgcn_wmma_f32_16x16x32_bf16(false, a, false, b, (short)0, c, false, false);
  }
  int colg = n0 + wave * 16 + (lane & 15);
  int mbase = (lane >> 4) * 8;
#pragma unroll
  for (int r = 0; r < 8; ++r) {
    int m = mbase + r;
    out[(size_t)(t0 + m) * HH + colg] = c[r] * sgs[m];
  }
}

// ---------------- routed GEMM1: hmid = silu(Xe@Wg[e]) * (Xe@Wu[e]), gathered rows ----------------
__global__ __launch_bounds__(256)
void routed_gemm1(const __bf16* __restrict__ xb, const float* __restrict__ w_gate,
                  const float* __restrict__ w_up, const int* __restrict__ rowtok,
                  const int* __restrict__ offs, __bf16* __restrict__ hmid) {
  __shared__ __bf16 As[BM * BKP];
  __shared__ __bf16 Bg[BN * BKP];
  __shared__ __bf16 Bu[BN * BKP];
  int mt = blockIdx.x;
  int ntiles = offs[EE] >> 4;
  if (mt >= ntiles) return;
  int row0 = mt * 16;
  int e = 0;
#pragma unroll
  for (int i = 1; i < EE; ++i) if (row0 >= offs[i]) e = i;
  int n0 = blockIdx.y * BN;
  int tid = threadIdx.x, lane = tid & 31, wave = tid >> 5;
  int lrow = lane & 15;
  int am = tid >> 4, ak = (tid & 15) * 2;
  int bn = tid & 127, bkh = (tid >> 7) * 16;
  const float* bg_base = w_gate + (size_t)e * HH * II;
  const float* bu_base = w_up + (size_t)e * HH * II;
  const __bf16* arow = xb + (size_t)rowtok[row0 + am] * HH + ak;  // gathered row
  v8f cg = {}; v8f cu = {};
  unsigned aReg; float bgR[16], buR[16];
  aReg = *(const unsigned*)(arow);
  B_LOAD(bgR, bg_base, II, 0)
  B_LOAD(buR, bu_base, II, 0)
  for (int k0 = 0; k0 < HH; k0 += BK) {
    __syncthreads();
    *(unsigned*)(&As[am * BKP + ak]) = aReg;
    B_STORE(bgR, Bg)
    B_STORE(buR, Bu)
    __syncthreads();
    int kn = k0 + BK;
    if (kn < HH) {
      aReg = *(const unsigned*)(arow + kn);
      B_LOAD(bgR, bg_base, II, kn)
      B_LOAD(buR, bu_base, II, kn)
    }
    v16bf a = ld_frag(&As[lrow * BKP], lane);
    v16bf fg = ld_frag(&Bg[(wave * 16 + lrow) * BKP], lane);
    v16bf fu = ld_frag(&Bu[(wave * 16 + lrow) * BKP], lane);
    cg = __builtin_amdgcn_wmma_f32_16x16x32_bf16(false, a, false, fg, (short)0, cg, false, false);
    cu = __builtin_amdgcn_wmma_f32_16x16x32_bf16(false, a, false, fu, (short)0, cu, false, false);
  }
  int colg = n0 + wave * 16 + (lane & 15);
  int mbase = (lane >> 4) * 8;
#pragma unroll
  for (int r = 0; r < 8; ++r) {
    float h = silu(cg[r]) * cu[r];
    hmid[(size_t)(row0 + mbase + r) * II + colg] = (__bf16)h;
  }
}

// ---------------- routed GEMM2: out += roww * (hmid @ Wd[e]), atomic combine ----------------
__global__ __launch_bounds__(256)
void routed_gemm2(const __bf16* __restrict__ hmid, const float* __restrict__ w_down,
                  const int* __restrict__ rowtok, const float* __restrict__ roww,
                  const int* __restrict__ offs, float* __restrict__ out) {
  __shared__ __bf16 As[BM * BKP];
  __shared__ __bf16 Bs[BN * BKP];
  __shared__ int rtok[BM];
  __shared__ float wrw[BM];
  int mt = blockIdx.x;
  int ntiles = offs[EE] >> 4;
  if (mt >= ntiles) return;
  int row0 = mt * 16;
  int e = 0;
#pragma unroll
  for (int i = 1; i < EE; ++i) if (row0 >= offs[i]) e = i;
  int n0 = blockIdx.y * BN;
  int tid = threadIdx.x, lane = tid & 31, wave = tid >> 5;
  int lrow = lane & 15;
  int am = tid >> 4, ak = (tid & 15) * 2;
  int bn = tid & 127, bkh = (tid >> 7) * 16;
  if (tid < BM) { rtok[tid] = rowtok[row0 + tid]; wrw[tid] = roww[row0 + tid]; }
  const float* bd_base = w_down + (size_t)e * II * HH;
  const __bf16* arow = hmid + (size_t)(row0 + am) * II + ak;
  v8f c = {};
  unsigned aReg; float bR[16];
  aReg = *(const unsigned*)(arow);
  B_LOAD(bR, bd_base, HH, 0)
  for (int k0 = 0; k0 < II; k0 += BK) {
    __syncthreads();
    *(unsigned*)(&As[am * BKP + ak]) = aReg;
    B_STORE(bR, Bs)
    __syncthreads();
    int kn = k0 + BK;
    if (kn < II) {
      aReg = *(const unsigned*)(arow + kn);
      B_LOAD(bR, bd_base, HH, kn)
    }
    v16bf a = ld_frag(&As[lrow * BKP], lane);
    v16bf b = ld_frag(&Bs[(wave * 16 + lrow) * BKP], lane);
    c = __builtin_amdgcn_wmma_f32_16x16x32_bf16(false, a, false, b, (short)0, c, false, false);
  }
  int colg = n0 + wave * 16 + (lane & 15);
  int mbase = (lane >> 4) * 8;
#pragma unroll
  for (int r = 0; r < 8; ++r) {
    int m = mbase + r;
    atomicAdd(&out[(size_t)rtok[m] * HH + colg], c[r] * wrw[m]);
  }
}

// ---------------- workspace layout ----------------
static constexpr size_t al256(size_t x) { return (x + 255) & ~(size_t)255; }
static constexpr size_t XB_OFF   = 0;
static constexpr size_t TOPI_OFF = al256(XB_OFF + (size_t)TT * HH * 2);
static constexpr size_t TOPW_OFF = al256(TOPI_OFF + (size_t)TT * 2 * 4);
static constexpr size_t SG_OFF   = al256(TOPW_OFF + (size_t)TT * 2 * 4);
static constexpr size_t CNT_OFF  = al256(SG_OFF + (size_t)TT * 4);
static constexpr size_t OFFS_OFF = al256(CNT_OFF + EE * 4);
static constexpr size_t CUR_OFF  = al256(OFFS_OFF + (EE + 1) * 4);
static constexpr size_t RTOK_OFF = al256(CUR_OFF + EE * 4);
static constexpr size_t RW_OFF   = al256(RTOK_OFF + (size_t)MAXROWS * 4);
static constexpr size_t HMID_OFF = al256(RW_OFF + (size_t)MAXROWS * 4);
static constexpr size_t SMID_OFF = al256(HMID_OFF + (size_t)MAXROWS * II * 2);
// total = SMID_OFF + T*SI*2  (~34 MB)

extern "C" void kernel_launch(void* const* d_in, const int* in_sizes, int n_in,
                              void* d_out, int out_size, void* d_ws, size_t ws_size,
                              hipStream_t stream) {
  const float* x        = (const float*)d_in[0];
  const float* w_router = (const float*)d_in[1];
  const float* w_shgate = (const float*)d_in[2];
  const float* w_gate   = (const float*)d_in[3];
  const float* w_up     = (const float*)d_in[4];
  const float* w_down   = (const float*)d_in[5];
  const float* w_sh_g   = (const float*)d_in[6];
  const float* w_sh_u   = (const float*)d_in[7];
  const float* w_sh_d   = (const float*)d_in[8];
  float* out = (float*)d_out;

  char* ws = (char*)d_ws;
  __bf16* xb    = (__bf16*)(ws + XB_OFF);
  int*    topi  = (int*)(ws + TOPI_OFF);
  float*  topw  = (float*)(ws + TOPW_OFF);
  float*  sg    = (float*)(ws + SG_OFF);
  int*    cnt   = (int*)(ws + CNT_OFF);
  int*    offs  = (int*)(ws + OFFS_OFF);
  int*    cur   = (int*)(ws + CUR_OFF);
  int*    rtok  = (int*)(ws + RTOK_OFF);
  float*  rww   = (float*)(ws + RW_OFF);
  __bf16* hmid  = (__bf16*)(ws + HMID_OFF);
  __bf16* smid  = (__bf16*)(ws + SMID_OFF);

  init_kernel<<<1, 64, 0, stream>>>(cnt, cur);
  cvt_bf16_kernel<<<(TT * HH / 4 + 255) / 256, 256, 0, stream>>>(x, xb, TT * HH / 4);
  router_kernel<<<TT / 8, 256, 0, stream>>>(x, w_router, w_shgate, topi, topw, sg);
  count_kernel<<<(TT * 2) / 256, 256, 0, stream>>>(topi, cnt);
  offsets_kernel<<<1, 64, 0, stream>>>(cnt, offs, rtok, rww);
  place_kernel<<<(TT * 2) / 256, 256, 0, stream>>>(topi, topw, offs, cur, rtok, rww);

  // shared expert (writes out with plain stores -> initializes d_out)
  shared_gemm1<<<dim3(TT / BM, SSI / BN), 256, 0, stream>>>(xb, w_sh_g, w_sh_u, smid);
  shared_gemm2<<<dim3(TT / BM, HH / BN), 256, 0, stream>>>(smid, w_sh_d, sg, out);

  // routed experts (atomic-add on top of shared-expert output)
  routed_gemm1<<<dim3(MAXTILES, II / BN), 256, 0, stream>>>(xb, w_gate, w_up, rtok, offs, hmid);
  routed_gemm2<<<dim3(MAXTILES, HH / BN), 256, 0, stream>>>(hmid, w_down, rtok, rww, offs, out);
}
